// Attention_13932873909162
// MI455X (gfx1250) — matrix operand order repaired and checked by reading, not compiled, not run on hardware
//
#include <hip/hip_runtime.h>
#include <hip/hip_bf16.h>
#include <math.h>

// Problem constants (match reference)
#define B_    2
#define S_    2048
#define DIM_  4096
#define NH_   32
#define NKV_  8
#define HD_   128      // head dim
#define QKVN_ 6144     // 4096 q + 1024 k + 1024 v

typedef __bf16 bf16_t;
typedef bf16_t v16bf __attribute__((ext_vector_type(16)));
typedef float  v8f   __attribute__((ext_vector_type(8)));
typedef int    v4i   __attribute__((ext_vector_type(4)));

union FragBF { uint4 u[2]; v16bf v; };

static __device__ __forceinline__ bf16_t f2bf(float f) { return (bf16_t)f; }
static __device__ __forceinline__ float  bf2f(bf16_t b) { return (float)b; }

#if __has_builtin(__builtin_amdgcn_global_load_async_to_lds_b128)
#define HAS_ASYNC_LDS 1
#else
#define HAS_ASYNC_LDS 0
#endif

#if HAS_ASYNC_LDS
// One 16-byte async global->LDS copy per lane (ASYNCcnt-tracked, no VGPR bounce).
// Builtin signature (from compiler diagnostic): (int4 AS1*, int4 AS3*, imm, imm).
static __device__ __forceinline__ void async_ld_b128(const bf16_t* gsrc, bf16_t* ldst) {
  __builtin_amdgcn_global_load_async_to_lds_b128(
      (__attribute__((address_space(1))) v4i*)gsrc,
      (__attribute__((address_space(3))) v4i*)ldst,
      0, 0);
}
static __device__ __forceinline__ void wait_async0() {
#if __has_builtin(__builtin_amdgcn_s_wait_asynccnt)
  __builtin_amdgcn_s_wait_asynccnt(0);
#else
  asm volatile("s_wait_asynccnt 0" ::: "memory");
#endif
}
#endif

// ---------------------------------------------------------------------------
// f32 -> bf16 elementwise convert
// ---------------------------------------------------------------------------
__global__ void cvt_f32_bf16(const float* __restrict__ src, bf16_t* __restrict__ dst, int n) {
  int i = blockIdx.x * blockDim.x + threadIdx.x;
  if (i < n) dst[i] = f2bf(src[i]);
}

// ---------------------------------------------------------------------------
// Transpose-convert: src [K=4096 rows][srcCols] f32 -> dst[(nOff+n)*4096 + k] bf16
// 64x64 tiles via LDS.
// ---------------------------------------------------------------------------
__global__ __launch_bounds__(256) void tconv(const float* __restrict__ src,
                                             bf16_t* __restrict__ dst,
                                             int srcCols, int nOff) {
  __shared__ bf16_t tile[64][65];
  const int n0 = blockIdx.x * 64, k0 = blockIdx.y * 64;
  const int t = threadIdx.x;
  const int nc = t & 63, r0 = (t >> 6) * 16;
#pragma unroll
  for (int i = 0; i < 16; i++)
    tile[r0 + i][nc] = f2bf(src[(size_t)(k0 + r0 + i) * srcCols + n0 + nc]);
  __syncthreads();
  const int kc = t & 63;
#pragma unroll
  for (int i = 0; i < 16; i++)
    dst[(size_t)(nOff + n0 + r0 + i) * 4096 + k0 + kc] = tile[kc][r0 + i];
}

// ---------------------------------------------------------------------------
// bf16 GEMM: C[M,N] = A[M,K] * Bt[N,K]^T, f32 accumulate via WMMA.
// Block: 256 threads (8 waves), tile 128x128, K-step 32, double-buffered LDS.
// Staging uses GLOBAL_LOAD_ASYNC_TO_LDS_B128 when available (prefetch next
// K-tile during compute; one s_wait_asynccnt 0 per step).
// Wave tile 32(M) x 64(N) = 2x4 WMMA C tiles, 8 WMMAs per wave per K-step.
// ---------------------------------------------------------------------------
template <bool OUT_F32>
__global__ __launch_bounds__(256) void gemm_bf16(const bf16_t* __restrict__ A,
                                                 const bf16_t* __restrict__ Bt,
                                                 void* __restrict__ Cout,
                                                 int M, int N, int K) {
  __shared__ __align__(16) bf16_t As[2][128 * 40];
  __shared__ __align__(16) bf16_t Bs[2][128 * 40];
  const int t = threadIdx.x;
  const int l = t & 31, wid = t >> 5;
  const int waveM = wid & 3, waveN = wid >> 2;
  const int m0 = blockIdx.y * 128, n0 = blockIdx.x * 128;
  const int mbase = waveM * 32, nbase = waveN * 64;
  const int nloc = l & 15;
  const int aoff = (l < 16) ? 0 : 8;    // A-frag K base (halfs)
  const int boff = (l < 16) ? 0 : 8;    // B-frag K base (halfs)

  v8f c[2][4];
#pragma unroll
  for (int mi = 0; mi < 2; mi++)
#pragma unroll
    for (int ni = 0; ni < 4; ni++)
#pragma unroll
      for (int r = 0; r < 8; r++) c[mi][ni][r] = 0.0f;

  const int srow = t >> 1;             // 0..127
  const int skh = (t & 1) << 4;        // 0 or 16 halfs

  // Stage one 128x32 A tile + 128x32 B tile into LDS buffer `buf`.
  auto stage = [&](int k0, int buf) {
    const bf16_t* ga = A + (size_t)(m0 + srow) * K + k0 + skh;
    const bf16_t* gb = Bt + (size_t)(n0 + srow) * K + k0 + skh;
    bf16_t* la = &As[buf][srow * 40 + skh];
    bf16_t* lb = &Bs[buf][srow * 40 + skh];
#if HAS_ASYNC_LDS
    async_ld_b128(ga,     la);
    async_ld_b128(ga + 8, la + 8);
    async_ld_b128(gb,     lb);
    async_ld_b128(gb + 8, lb + 8);
#else
    uint4 a0 = ((const uint4*)ga)[0], a1 = ((const uint4*)ga)[1];
    uint4 b0 = ((const uint4*)gb)[0], b1 = ((const uint4*)gb)[1];
    ((uint4*)la)[0] = a0; ((uint4*)la)[1] = a1;
    ((uint4*)lb)[0] = b0; ((uint4*)lb)[1] = b1;
#endif
  };

  const int nsteps = K >> 5;
  stage(0, 0);

  for (int i = 0; i < nsteps; i++) {
    const int buf = i & 1;
#if HAS_ASYNC_LDS
    wait_async0();              // this buffer's async loads have landed in LDS
#endif
    __syncthreads();            // visible to all waves
    if (i + 1 < nsteps) stage((i + 1) << 5, buf ^ 1);   // prefetch next tile

    FragBF af[2];
#pragma unroll
    for (int mi = 0; mi < 2; mi++) {
      const uint4* p = (const uint4*)&As[buf][(mbase + mi * 16 + nloc) * 40 + aoff];
      af[mi].u[0] = p[0];   // K halfs [aoff, aoff+8)
      af[mi].u[1] = p[2];   // K halfs [aoff+16, aoff+24)
    }
#pragma unroll
    for (int ni = 0; ni < 4; ni++) {
      FragBF bf;
      const uint4* p = (const uint4*)&Bs[buf][(nbase + ni * 16 + nloc) * 40 + boff];
      bf.u[0] = p[0]; bf.u[1] = p[2];
      c[0][ni] = __builtin_amdgcn_wmma_f32_16x16x32_bf16(false, af[0].v, false, bf.v,
                                                         (short)0, c[0][ni], false, false);
      c[1][ni] = __builtin_amdgcn_wmma_f32_16x16x32_bf16(false, af[1].v, false, bf.v,
                                                         (short)0, c[1][ni], false, false);
    }
    __syncthreads();            // all reads done before this buffer is overwritten
  }

  const int hi8 = (l >= 16) ? 8 : 0;
#pragma unroll
  for (int mi = 0; mi < 2; mi++)
#pragma unroll
    for (int ni = 0; ni < 4; ni++)
#pragma unroll
      for (int r = 0; r < 8; r++) {
        int mg = m0 + mbase + mi * 16 + r + hi8;
        int ng = n0 + nbase + ni * 16 + nloc;
        float v = c[mi][ni][r];
        if (OUT_F32) ((float*)Cout)[(size_t)mg * N + ng] = v;
        else         ((bf16_t*)Cout)[(size_t)mg * N + ng] = f2bf(v);
      }
}

// ---------------------------------------------------------------------------
// RoPE + head split: qkv [tok,6144] bf16 -> qr [b,32,s,128], kr [b,8,s,128],
// vt [b,8,128,s] (V transposed for PV WMMA B-fragments). One thread per (tok,h,pair).
// ---------------------------------------------------------------------------
__global__ void rope_split(const bf16_t* __restrict__ qkv, const float* __restrict__ fc,
                           bf16_t* __restrict__ qr, bf16_t* __restrict__ kr,
                           bf16_t* __restrict__ vt) {
  int idx = blockIdx.x * blockDim.x + threadIdx.x;   // B*S*32*64 threads
  int p   = idx & 63;
  int h   = (idx >> 6) & 31;
  int tok = idx >> 11;              // 0..B*S-1
  int b   = tok >> 11;              // S = 2048
  int s   = tok & 2047;
  float cs = fc[(s * 64 + p) * 2 + 0];
  float sn = fc[(s * 64 + p) * 2 + 1];
  const bf16_t* row = qkv + (size_t)tok * QKVN_;

  float q0 = bf2f(row[h * 128 + 2 * p]);
  float q1 = bf2f(row[h * 128 + 2 * p + 1]);
  size_t qo = ((size_t)(b * 32 + h) * S_ + s) * 128 + 2 * p;
  qr[qo]     = f2bf(q0 * cs - q1 * sn);
  qr[qo + 1] = f2bf(q1 * cs + q0 * sn);

  if (h < 8) {
    float k0 = bf2f(row[4096 + h * 128 + 2 * p]);
    float k1 = bf2f(row[4096 + h * 128 + 2 * p + 1]);
    size_t ko = ((size_t)(b * 8 + h) * S_ + s) * 128 + 2 * p;
    kr[ko]     = f2bf(k0 * cs - k1 * sn);
    kr[ko + 1] = f2bf(k1 * cs + k0 * sn);
    bf16_t v0 = row[5120 + h * 128 + 2 * p];
    bf16_t v1 = row[5120 + h * 128 + 2 * p + 1];
    size_t vo = ((size_t)(b * 8 + h) * 128 + 2 * p) * S_ + s;
    vt[vo]        = v0;
    vt[vo + S_]   = v1;
  }
}

// ---------------------------------------------------------------------------
// Flash attention, causal, GQA (4 q heads per kv head).
// One wave per 16-row Q tile; key blocks of 32; online softmax.
// Block = 128 threads (4 waves, 4 consecutive Q tiles of the same head).
// ---------------------------------------------------------------------------
__global__ __launch_bounds__(128) void attn_kernel(const bf16_t* __restrict__ qr,
                                                   const bf16_t* __restrict__ kr,
                                                   const bf16_t* __restrict__ vt,
                                                   bf16_t* __restrict__ ao) {
  __shared__ __align__(16) bf16_t Plds[4][16 * 40];
  const int l = threadIdx.x & 31, wid = threadIdx.x >> 5;
  const int h = blockIdx.y, b = blockIdx.z;
  const int q0 = (blockIdx.x * 4 + wid) * 16;
  const int hkv = h >> 2;
  const bf16_t* qb = qr + ((size_t)(b * 32 + h) * S_ + q0) * 128;
  const bf16_t* kb_ = kr + (size_t)(b * 8 + hkv) * S_ * 128;
  const bf16_t* vb = vt + (size_t)(b * 8 + hkv) * 128 * S_;
  const int nloc = l & 15;
  const int hi   = l >> 4;
  const int hi8  = hi * 8;
  const int aoff = hi ? 8 : 0;    // A-frag K base
  const int boff = hi ? 8 : 0;    // B-frag K base

  // Q fragments: 4 chunks of K=32 over head dim 128 (A layout, rows q0..q0+15)
  FragBF qf[4];
#pragma unroll
  for (int c2 = 0; c2 < 4; c2++) {
    const uint4* p = (const uint4*)(qb + (size_t)nloc * 128 + c2 * 32 + aoff);
    qf[c2].u[0] = p[0];
    qf[c2].u[1] = p[2];
  }

  v8f o[8];
  float rmax[8], rsum[8];
#pragma unroll
  for (int nd = 0; nd < 8; nd++)
#pragma unroll
    for (int r = 0; r < 8; r++) o[nd][r] = 0.0f;
#pragma unroll
  for (int r = 0; r < 8; r++) { rmax[r] = -INFINITY; rsum[r] = 0.0f; }

  const float SCALE = 0.08838834764831845f;   // 1/sqrt(128)
  bf16_t* Pw = Plds[wid];
  const int nkb = (q0 + 16 + 31) >> 5;

  for (int kb = 0; kb < nkb; kb++) {
    const int kb0 = kb * 32;
    // scores: two 16x16 tiles (keys kb0..+15, kb0+16..+31)
    v8f st[2];
#pragma unroll
    for (int tt = 0; tt < 2; tt++) {
      v8f acc;
#pragma unroll
      for (int r = 0; r < 8; r++) acc[r] = 0.0f;
#pragma unroll
      for (int c2 = 0; c2 < 4; c2++) {
        FragBF kf;
        const uint4* p =
            (const uint4*)(kb_ + (size_t)(kb0 + tt * 16 + nloc) * 128 + c2 * 32 + boff);
        kf.u[0] = p[0]; kf.u[1] = p[2];
        acc = __builtin_amdgcn_wmma_f32_16x16x32_bf16(false, qf[c2].v, false, kf.v,
                                                      (short)0, acc, false, false);
      }
      st[tt] = acc;
    }

    // online softmax update (per row m = r + hi8; reduce over 16-lane halves)
    float pr0[8], pr1[8];
#pragma unroll
    for (int r = 0; r < 8; r++) {
      const int mg = q0 + r + hi8;
      float s0 = st[0][r] * SCALE;
      float s1 = st[1][r] * SCALE;
      if (kb0 + nloc > mg)      s0 += -1e9f;
      if (kb0 + 16 + nloc > mg) s1 += -1e9f;
      float mx = fmaxf(s0, s1);
      mx = fmaxf(mx, __shfl_xor(mx, 1, 32));
      mx = fmaxf(mx, __shfl_xor(mx, 2, 32));
      mx = fmaxf(mx, __shfl_xor(mx, 4, 32));
      mx = fmaxf(mx, __shfl_xor(mx, 8, 32));
      float nm = fmaxf(rmax[r], mx);
      float sc = __expf(rmax[r] - nm);
      float p0 = __expf(s0 - nm);
      float p1 = __expf(s1 - nm);
      float bs = p0 + p1;
      bs += __shfl_xor(bs, 1, 32);
      bs += __shfl_xor(bs, 2, 32);
      bs += __shfl_xor(bs, 4, 32);
      bs += __shfl_xor(bs, 8, 32);
      rsum[r] = rsum[r] * sc + bs;
      rmax[r] = nm;
#pragma unroll
      for (int nd = 0; nd < 8; nd++) o[nd][r] *= sc;
      pr0[r] = p0; pr1[r] = p1;
    }

    // P tile (C layout) -> LDS row-major [m][key] -> reload as A fragment
#pragma unroll
    for (int r = 0; r < 8; r++) {
      Pw[(r + hi8) * 40 + nloc]      = f2bf(pr0[r]);
      Pw[(r + hi8) * 40 + 16 + nloc] = f2bf(pr1[r]);
    }
    asm volatile("s_wait_dscnt 0" ::: "memory");
    FragBF pf;
    {
      const uint4* p = (const uint4*)&Pw[nloc * 40 + aoff];
      pf.u[0] = p[0];
      pf.u[1] = p[2];
    }

    // O += P x V  (V fragments from transposed vt: [d][s])
#pragma unroll
    for (int nd = 0; nd < 8; nd++) {
      FragBF vf;
      const uint4* p = (const uint4*)(vb + (size_t)(nd * 16 + nloc) * S_ + kb0 + boff);
      vf.u[0] = p[0]; vf.u[1] = p[2];
      o[nd] = __builtin_amdgcn_wmma_f32_16x16x32_bf16(false, pf.v, false, vf.v,
                                                      (short)0, o[nd], false, false);
    }
  }

  // epilogue: normalize and store bf16 to ao[token, h*128 + d]
#pragma unroll
  for (int r = 0; r < 8; r++) {
    float inv = 1.0f / rsum[r];
    size_t base = ((size_t)(b * S_) + q0 + r + hi8) * 4096 + h * 128;
#pragma unroll
    for (int nd = 0; nd < 8; nd++)
      ao[base + nd * 16 + nloc] = f2bf(o[nd][r] * inv);
  }
}

// ---------------------------------------------------------------------------
// Host-side orchestration. Inputs: x, start_pos, freqs_cis, mask, wq, wk, wv, wo.
// Workspace layout (bytes), total 176 MB with safe aliasing:
//   [0,32M)    xb  (x bf16)            -> reused as ao after GEMM1
//   [32M,80M)  wqkv_t bf16 [6144,4096] -> first 32MB reused as wo_t after GEMM1
//   [80M,128M) qkv bf16 [4096,6144]
//   [128M,160M) qr ; [160M,168M) kr ; [168M,176M) vt
// ---------------------------------------------------------------------------
extern "C" void kernel_launch(void* const* d_in, const int* in_sizes, int n_in,
                              void* d_out, int out_size, void* d_ws, size_t ws_size,
                              hipStream_t stream) {
  const float* x  = (const float*)d_in[0];
  const float* fc = (const float*)d_in[2];
  const float* wq = (const float*)d_in[4];
  const float* wk = (const float*)d_in[5];
  const float* wv = (const float*)d_in[6];
  const float* wo = (const float*)d_in[7];

  char* ws = (char*)d_ws;
  bf16_t* xb     = (bf16_t*)(ws);
  bf16_t* wqkv_t = (bf16_t*)(ws + ((size_t)32 << 20));
  bf16_t* qkv    = (bf16_t*)(ws + ((size_t)80 << 20));
  bf16_t* qr     = (bf16_t*)(ws + ((size_t)128 << 20));
  bf16_t* kr     = (bf16_t*)(ws + ((size_t)160 << 20));
  bf16_t* vt     = (bf16_t*)(ws + ((size_t)168 << 20));
  bf16_t* ao     = xb;        // alias: xb dead after GEMM1
  bf16_t* wo_t   = wqkv_t;    // alias: wqkv_t dead after GEMM1

  const int ntok = B_ * S_;   // 4096

  cvt_f32_bf16<<<(ntok * DIM_) / 256, 256, 0, stream>>>(x, xb, ntok * DIM_);
  tconv<<<dim3(4096 / 64, 4096 / 64), 256, 0, stream>>>(wq, wqkv_t, 4096, 0);
  tconv<<<dim3(1024 / 64, 4096 / 64), 256, 0, stream>>>(wk, wqkv_t, 1024, 4096);
  tconv<<<dim3(1024 / 64, 4096 / 64), 256, 0, stream>>>(wv, wqkv_t, 1024, 5120);

  gemm_bf16<false><<<dim3(QKVN_ / 128, ntok / 128), 256, 0, stream>>>(
      xb, wqkv_t, qkv, ntok, QKVN_, DIM_);

  rope_split<<<(ntok * 32 * 64) / 256, 256, 0, stream>>>(qkv, fc, qr, kr, vt);
  tconv<<<dim3(4096 / 64, 4096 / 64), 256, 0, stream>>>(wo, wo_t, 4096, 0);

  attn_kernel<<<dim3(S_ / 64, NH_, B_), 128, 0, stream>>>(qr, kr, vt, ao);

  gemm_bf16<true><<<dim3(DIM_ / 128, ntok / 128), 256, 0, stream>>>(
      ao, wo_t, d_out, ntok, DIM_, DIM_);
}
